// Model_90168543412725
// MI455X (gfx1250) — matrix operand order repaired and checked by reading, not compiled, not run on hardware
//
#include <hip/hip_runtime.h>
#include <hip/hip_bf16.h>
#include <math.h>

// ---------------------------------------------------------------------------
// MI455X (gfx1250) implementation.
//  B=16, R=8, C=1023 cities, N=1024 nodes, H=64, M = B*N = 16384 rows.
//
//  * loop-invariant A[b,i,c,h] = presence_in * sigmoid(dist * w_emb_h) is
//    materialized once as bf16 (A1 2.1GB, A2 4.3GB); the 10 message-passing
//    iterations then stream A at HBM rate (~32GB -> ~1.4ms @ 23.3TB/s),
//    u[b] (256KB) stays L2-resident.
//  * dense GEMMs run on v_wmma_f32_16x16x32_bf16 with bf16 A operands
//    produced directly by the upstream kernels (no per-kstep cvt ALU);
//    weight panels staged into LDS by the Tensor Data Mover.
// ---------------------------------------------------------------------------

#define BB 16
#define RR 8
#define NC 1023
#define NN 1024
#define HHH 64
#define MM (BB*NN)          // 16384
#define NEGV 1.0e10f

typedef __attribute__((ext_vector_type(16))) __bf16 v16bf;
typedef __attribute__((ext_vector_type(8)))  float  v8f;
typedef __attribute__((ext_vector_type(8)))  unsigned v8u;
typedef __attribute__((ext_vector_type(4)))  unsigned v4u;
typedef __attribute__((ext_vector_type(8)))  int      v8i;
typedef __attribute__((ext_vector_type(4)))  int      v4i;

#if __has_builtin(__builtin_amdgcn_tensor_load_to_lds) && \
    __has_builtin(__builtin_amdgcn_s_wait_tensorcnt)
#define USE_TDM 1
#else
#define USE_TDM 0
#endif

__device__ __forceinline__ __bf16 f2bf(float f) {
  unsigned u = __builtin_bit_cast(unsigned, f);
  unsigned r = u + 0x7FFFu + ((u >> 16) & 1u);      // round-to-nearest-even
  unsigned short h = (unsigned short)(r >> 16);
  return __builtin_bit_cast(__bf16, h);
}
__device__ __forceinline__ unsigned short f2bfu(float f) {
  unsigned u = __builtin_bit_cast(unsigned, f);
  unsigned r = u + 0x7FFFu + ((u >> 16) & 1u);
  return (unsigned short)(r >> 16);
}
__device__ __forceinline__ float bf_lo(unsigned p) {
  return __builtin_bit_cast(float, p << 16);
}
__device__ __forceinline__ float bf_hi(unsigned p) {
  return __builtin_bit_cast(float, p & 0xFFFF0000u);
}

// ---------------------------------------------------------------------------
// x_a[b,n] = max_r x_a_state[b,r,n] * (assignment_prev + action)[b,r,n]
// ---------------------------------------------------------------------------
__global__ void xa_kernel(const float* __restrict__ xs,
                          const float* __restrict__ ap,
                          const float* __restrict__ ac,
                          float* __restrict__ xa) {
  int i = blockIdx.x * 256 + threadIdx.x;         // b*1024 + n
  int b = i >> 10, n = i & 1023;
  float m = -3.4e38f;
  #pragma unroll
  for (int r = 0; r < RR; ++r) {
    size_t o = ((size_t)(b * RR + r)) * NN + n;
    m = fmaxf(m, xs[o] * (ap[o] + ac[o]));
  }
  xa[i] = m;
}

// Xab[m,col] = x_a[m] * (col<64 ? W_x_1a[col] : W_x_1b[col-64])
__global__ void xbias_kernel(const float* __restrict__ xa,
                             const float* __restrict__ wxa,
                             const float* __restrict__ wxb,
                             float* __restrict__ Xab) {
  int idx = blockIdx.x * 256 + threadIdx.x;       // m*128 + col
  int m = idx >> 7, col = idx & 127;
  float w = (col < HHH) ? wxa[col] : wxb[col - HHH];
  Xab[idx] = xa[m] * w;
}

// pack two [K=64,64] f32 weights -> one TRANSPOSED bf16 panel out[n*64+k], n<128
__global__ void pack_cat_kernel(const float* __restrict__ Wa,
                                const float* __restrict__ Wb,
                                __bf16* __restrict__ outT) {
  int idx = blockIdx.x * 256 + threadIdx.x;       // k*128 + col
  int k = idx >> 7, col = idx & 127;
  float v = (col < HHH) ? Wa[k * HHH + col] : Wb[k * HHH + (col - HHH)];
  outT[col * HHH + k] = f2bf(v);                  // [n][k], K=64
}

// pack [128,128] f32 -> TRANSPOSED bf16 panel out[n*128+k]
__global__ void pack_kernel(const float* __restrict__ W, __bf16* __restrict__ outT) {
  int idx = blockIdx.x * 256 + threadIdx.x;       // k*128 + n
  int k = idx >> 7, n = idx & 127;
  outT[n * 128 + k] = f2bf(W[idx]);
}

// U[m,col] = col<64 ? u_a0[m,col] : u_b0[m,col-64]
__global__ void initU_kernel(const float* __restrict__ ua,
                             const float* __restrict__ ub,
                             float* __restrict__ U) {
  int idx = blockIdx.x * 256 + threadIdx.x;
  int m = idx >> 7, col = idx & 127;
  U[idx] = (col < HHH) ? ua[(size_t)m * HHH + col] : ub[(size_t)m * HHH + (col - HHH)];
}

// ---------------------------------------------------------------------------
// Presence MLP: logit[b,c,n] = relu(relu(edge@W1)@W2)/TAU masked; dT[b,n,c]=dist
// ---------------------------------------------------------------------------
__global__ void presence_kernel(const float* __restrict__ edge,
                                const float* __restrict__ avail,
                                const float* __restrict__ W1,   // [3,64]
                                const float* __restrict__ W2,   // [64,1]
                                float* __restrict__ logit,      // [B,C,N]
                                float* __restrict__ dT) {       // [B,N,C]
  int bc = blockIdx.x;                  // b*1023 + c
  int b = bc / NC, c = bc - b * NC;
  __shared__ float w1[3 * HHH];
  __shared__ float w2[HHH];
  if (threadIdx.x < 192) w1[threadIdx.x] = W1[threadIdx.x];
  if (threadIdx.x >= 192) w2[threadIdx.x - 192] = W2[threadIdx.x - 192];
  __syncthreads();

  for (int n = threadIdx.x; n < NN; n += 256) {
    const float* e = edge + ((size_t)bc * NN + n) * 3;
    float e0 = e[0], e1 = e[1], e2 = e[2];
    float h2 = 0.f;
    #pragma unroll
    for (int h = 0; h < HHH; ++h) {
      float v = fmaf(e0, w1[h], fmaf(e1, w1[HHH + h], e2 * w1[2 * HHH + h]));
      v = fmaxf(v, 0.f);
      h2 = fmaf(v, w2[h], h2);
    }
    h2 = fmaxf(h2, 0.f);                              // TAU == 1
    float mask = ((n == c) ? 0.f : 1.f) * avail[(size_t)b * NN + n];
    logit[(size_t)bc * NN + n] = h2 * mask - (1.f - mask) * NEGV;
    dT[((size_t)b * NN + n) * NC + c] = e0;           // transposed dist
  }
}

// softmax over n per (b,c) row, write presence_in transposed: Pin[b,n,c]
__global__ void softmax_kernel(const float* __restrict__ logit,
                               float* __restrict__ Pin) {
  int bc = blockIdx.x;
  int b = bc / NC, c = bc - b * NC;
  const float* row = logit + (size_t)bc * NN;
  __shared__ float sred[256];
  int tid = threadIdx.x;
  float v[4], m = -3.4e38f;
  #pragma unroll
  for (int k = 0; k < 4; ++k) { v[k] = row[tid + 256 * k]; m = fmaxf(m, v[k]); }
  sred[tid] = m; __syncthreads();
  for (int s = 128; s > 0; s >>= 1) {
    if (tid < s) sred[tid] = fmaxf(sred[tid], sred[tid + s]);
    __syncthreads();
  }
  float mx = sred[0]; __syncthreads();
  float e[4], sum = 0.f;
  #pragma unroll
  for (int k = 0; k < 4; ++k) { e[k] = __expf(v[k] - mx); sum += e[k]; }
  sred[tid] = sum; __syncthreads();
  for (int s = 128; s > 0; s >>= 1) {
    if (tid < s) sred[tid] += sred[tid + s];
    __syncthreads();
  }
  float inv = 1.f / sred[0];
  #pragma unroll
  for (int k = 0; k < 4; ++k) {
    int n = tid + 256 * k;
    Pin[((size_t)b * NN + n) * NC + c] = e[k] * inv;
  }
}

// ---------------------------------------------------------------------------
// Loop-invariant A tensors (bf16):
//   A1[b,i,c,h] = p * sigmoid(d * W_emb_1a[h]),  A2 with W_emb_2 (128 wide)
// ---------------------------------------------------------------------------
__global__ void buildA_kernel(const float* __restrict__ Pin,
                              const float* __restrict__ dT,
                              const float* __restrict__ we1,   // [64]
                              const float* __restrict__ we2,   // [128]
                              __bf16* __restrict__ A1,
                              __bf16* __restrict__ A2) {
  int bi = blockIdx.x;
  __shared__ float w1[HHH];
  __shared__ float w2[2 * HHH];
  if (threadIdx.x < HHH) w1[threadIdx.x] = we1[threadIdx.x];
  if (threadIdx.x < 2 * HHH) w2[threadIdx.x] = we2[threadIdx.x];
  __syncthreads();
  int h = threadIdx.x & 63;
  int cg = threadIdx.x >> 6;
  const float* pp = Pin + (size_t)bi * NC;
  const float* dp = dT + (size_t)bi * NC;
  for (int c = cg; c < NC; c += 4) {
    float p = pp[c], d = dp[c];
    size_t b1 = ((size_t)bi * NC + c) * HHH;
    size_t b2 = ((size_t)bi * NC + c) * (2 * HHH);
    A1[b1 + h]        = f2bf(p / (1.f + __expf(-d * w1[h])));
    A2[b2 + h]        = f2bf(p / (1.f + __expf(-d * w2[h])));
    A2[b2 + h + HHH]  = f2bf(p / (1.f + __expf(-d * w2[h + HHH])));
  }
}

// ---------------------------------------------------------------------------
// Einsum: L[b,i,h] = sum_c A[b,i,c,h] * U[(b*1024+c)*ustride + h]
// Each thread owns an h-quad: uint2 load (4x bf16 of A) + float4 of U per c.
// Output written as packed bf16 (feeds the WMMA GEMM A operand directly).
// ---------------------------------------------------------------------------
__global__ void einsum_kernel(const __bf16* __restrict__ A,
                              const float* __restrict__ U,
                              __bf16* __restrict__ L,
                              int Hd, int ustride) {
  int bi = blockIdx.x;                       // b*1024 + i
  int b = bi >> 10;
  int tid = threadIdx.x;
  int hq = Hd >> 2;                          // h-quads: 16 or 32
  int cpar = 256 / hq;                       // 16 or 8
  int hp = tid % hq;
  int cg = tid / hq;
  const uint2* Ap = (const uint2*)(A + (size_t)bi * NC * Hd) + hp;
  const float* Up = U + ((size_t)b << 10) * ustride + 4 * hp;
  float acc0 = 0.f, acc1 = 0.f, acc2 = 0.f, acc3 = 0.f;
  int niter = (NC - cg + cpar - 1) / cpar;
  for (int k = 0; k < niter; ++k) {
    int c = cg + k * cpar;
    if ((k & 7) == 0 && c + 8 * cpar < NC)
      __builtin_prefetch(&Ap[(size_t)(c + 8 * cpar) * hq], 0, 0);
    uint2 a4 = Ap[(size_t)c * hq];
    float4 uu = *(const float4*)(Up + (size_t)c * ustride);
    acc0 = fmaf(bf_lo(a4.x), uu.x, acc0);
    acc1 = fmaf(bf_hi(a4.x), uu.y, acc1);
    acc2 = fmaf(bf_lo(a4.y), uu.z, acc2);
    acc3 = fmaf(bf_hi(a4.y), uu.w, acc3);
  }
  __shared__ float red[1024];
  red[tid]       = acc0;
  red[256 + tid] = acc1;
  red[512 + tid] = acc2;
  red[768 + tid] = acc3;
  __syncthreads();
  if (cg == 0) {
    float s0 = acc0, s1 = acc1, s2 = acc2, s3 = acc3;
    for (int g = 1; g < cpar; ++g) {
      int o = g * hq + hp;
      s0 += red[o]; s1 += red[256 + o]; s2 += red[512 + o]; s3 += red[768 + o];
    }
    unsigned p0 = ((unsigned)f2bfu(s1) << 16) | (unsigned)f2bfu(s0);
    unsigned p1 = ((unsigned)f2bfu(s3) << 16) | (unsigned)f2bfu(s2);
    unsigned* Lw = (unsigned*)(L + (size_t)bi * Hd);
    Lw[2 * hp]     = p0;
    Lw[2 * hp + 1] = p1;
  }
}

// ---------------------------------------------------------------------------
// WMMA GEMM: D[M,128] = act( Abf[M,K] @ W[K,128] + bias ), K in {64,128}.
// Abf is bf16 (producer kernels emit it), so the A operand is two b128 loads
// per lane with zero conversion ALU. WbT (pre-transposed bf16 [n][k]) is
// staged into LDS by the Tensor Data Mover. Optional bf16 shadow output Dbf
// feeds downstream GEMMs. 8 waves/block, wave = 16 rows x 128 cols.
// ---------------------------------------------------------------------------
__global__ void __launch_bounds__(256) gemm_wmma_kernel(
    const __bf16* __restrict__ Abf, const __bf16* __restrict__ WbT,
    const float* __restrict__ bias, float* __restrict__ D,
    __bf16* __restrict__ Dbf,
    int K, int hasBias, int doRelu) {
  __shared__ __align__(128) __bf16 ldsW[128 * 128];   // [n][k]

#if USE_TDM
  if (threadIdx.x < 32) {                 // wave 0 issues the tensor DMA
    unsigned n8 = (unsigned)(K * 32);     // panel size in 8-byte units (=K*128*2/8)
    size_t ga = (size_t)WbT;
    unsigned lds_addr = (unsigned)(uintptr_t)(&ldsW[0]);   // low 32b = LDS offset
    // D# group0: count=1 | lds_addr | global_addr(57b) | type=2
    v4u g0 = { 1u, lds_addr, (unsigned)ga,
               (unsigned)((ga >> 32) & 0x1FFFFFFu) | 0x80000000u };
    // D# group1: data_size=3 (8B); 1-D tensor dim0=n8, dim1=1; tile dim0=n8;
    //            tensor_dim0_stride=n8.  (fields per ISA 8.4 bit layout)
    v8i g1 = { (int)(3u << 16),            // data_size
               (int)(n8 << 16),            // tensor_dim0[15:0] @ bits 63:48
               (int)(1u << 16),            // tensor_dim0 hi=0 | tensor_dim1=1
               (int)(n8 << 16),            // tensor_dim1 hi=0 | tile_dim0=n8
               0,                          // tile_dim1=0, tile_dim2=0
               (int)n8,                    // tensor_dim0_stride[31:0]
               0, 0 };
    v4i g2 = { 0, 0, 0, 0 };
    v4i g3 = { 0, 0, 0, 0 };
#if defined(__clang_major__) && (__clang_major__ >= 23)
    v8i g4 = { 0, 0, 0, 0, 0, 0, 0, 0 };
    __builtin_amdgcn_tensor_load_to_lds(g0, g1, g2, g3, g4, 0);
#else
    __builtin_amdgcn_tensor_load_to_lds(g0, g1, g2, g3, 0);
#endif
    __builtin_amdgcn_s_wait_tensorcnt(0);
  }
  __syncthreads();
#else
  for (int idx = threadIdx.x; idx < K * 128; idx += 256)
    ldsW[idx] = WbT[idx];
  __syncthreads();
#endif

  const int wave = threadIdx.x >> 5;
  const int lane = threadIdx.x & 31;
  const int r    = lane & 15;
  const int hi   = lane >> 4;                    // lane half
  const int m0   = blockIdx.x * 128 + wave * 16;

  v8f acc[8];
  #pragma unroll
  for (int t = 0; t < 8; ++t) {
    #pragma unroll
    for (int e = 0; e < 8; ++e) acc[t][e] = 0.f;
  }

  const int aoff = hi * 8;    // A: lanes 16-31 hold K 8..15 / 24..31
  const int boff = hi * 8;    // B: lanes 16-31 hold K 8..15 / 24..31

  for (int k0 = 0; k0 < K; k0 += 32) {
    // A operand: bf16 row fragment {K off+0..7, off+16..23} = two b128 loads
    const __bf16* arow = Abf + (size_t)(m0 + r) * K + k0 + aoff;
    uint4 q0 = *(const uint4*)(arow);
    uint4 q1 = *(const uint4*)(arow + 16);
    v8u tu;
    tu[0] = q0.x; tu[1] = q0.y; tu[2] = q0.z; tu[3] = q0.w;
    tu[4] = q1.x; tu[5] = q1.y; tu[6] = q1.z; tu[7] = q1.w;
    v16bf a = __builtin_bit_cast(v16bf, tu);

    // batch all 8 B-operand fetches, then issue the WMMAs back-to-back
    v16bf bvs[8];
    #pragma unroll
    for (int t = 0; t < 8; ++t) {
      const __bf16* bp = &ldsW[(t * 16 + r) * K + k0 + boff];
      #pragma unroll
      for (int j = 0; j < 16; ++j) bvs[t][j] = bp[j < 8 ? j : j + 8];
    }
    #pragma unroll
    for (int t = 0; t < 8; ++t)
      acc[t] = __builtin_amdgcn_wmma_f32_16x16x32_bf16(
          false, a, false, bvs[t], (short)0, acc[t], false, false);
  }

  #pragma unroll
  for (int t = 0; t < 8; ++t) {
    #pragma unroll
    for (int rr = 0; rr < 8; ++rr) {
      int row = m0 + rr + hi * 8;                // C/D layout: VGPR rr -> M rr / rr+8
      int col = t * 16 + r;
      float v = acc[t][rr];
      size_t o = (size_t)row * 128 + col;
      if (hasBias) v += bias[o];
      if (doRelu) v = fmaxf(v, 0.f);
      D[o] = v;
      if (Dbf) Dbf[o] = f2bf(v);
    }
  }
}

// Q[b] = sum_{n,h} G[b,n,h] * W_Q[h]
__global__ void q_kernel(const float* __restrict__ G,
                         const float* __restrict__ WQ,
                         float* __restrict__ out) {
  int b = blockIdx.x, tid = threadIdx.x;
  __shared__ float wq[128];
  __shared__ float red[256];
  if (tid < 128) wq[tid] = WQ[tid];
  __syncthreads();
  float acc = 0.f;
  const float* g = G + (size_t)b * NN * 128;
  for (int j = tid; j < NN * 128; j += 256) acc = fmaf(g[j], wq[j & 127], acc);
  red[tid] = acc; __syncthreads();
  for (int s = 128; s > 0; s >>= 1) {
    if (tid < s) red[tid] += red[tid + s];
    __syncthreads();
  }
  if (tid == 0) out[b] = red[0];
}

// ---------------------------------------------------------------------------
extern "C" void kernel_launch(void* const* d_in, const int* in_sizes, int n_in,
                              void* d_out, int out_size, void* d_ws, size_t ws_size,
                              hipStream_t stream) {
  const float* x_a_state = (const float*)d_in[0];
  const float* assn_prev = (const float*)d_in[1];
  const float* action    = (const float*)d_in[2];
  const float* edge      = (const float*)d_in[3];
  const float* avail     = (const float*)d_in[4];
  const float* u_a0      = (const float*)d_in[5];
  const float* u_b0      = (const float*)d_in[6];
  const float* gamma0    = (const float*)d_in[7];
  const float* W1p       = (const float*)d_in[8];
  const float* W2p       = (const float*)d_in[9];
  const float* W_x_1a    = (const float*)d_in[10];
  const float* W_emb_1a  = (const float*)d_in[11];
  const float* W_l_1a    = (const float*)d_in[12];
  const float* W_x_1b    = (const float*)d_in[13];
  // d_in[14] = W_emb_1b: dead in reference (u_b update reuses l_a)
  const float* W_l_1b    = (const float*)d_in[15];
  const float* W_x_2     = (const float*)d_in[16];
  const float* W_emb_2   = (const float*)d_in[17];
  const float* W_l_2     = (const float*)d_in[18];
  const float* W_Q       = (const float*)d_in[19];

  // workspace carve-out (~6.7 GB; A1/A2 dominate by design)
  char* ws = (char*)d_ws;
  size_t off = 0;
  auto take = [&](size_t bytes) -> char* {
    off = (off + 255) & ~(size_t)255;
    char* p = ws + off; off += bytes; return p;
  };
  __bf16* A1   = (__bf16*)take((size_t)MM * NC * HHH * 2);       // 2.14 GB
  __bf16* A2   = (__bf16*)take((size_t)MM * NC * 2 * HHH * 2);   // 4.29 GB
  float* logit = (float*)take((size_t)BB * NC * NN * 4);
  float* Pin   = (float*)take((size_t)MM * NC * 4);
  float* dT    = (float*)take((size_t)MM * NC * 4);
  float* U     = (float*)take((size_t)MM * 128 * 4);
  float* G     = (float*)take((size_t)MM * 128 * 4);
  float* Xab   = (float*)take((size_t)MM * 128 * 4);
  float* X2    = (float*)take((size_t)MM * 128 * 4);
  float* xa    = (float*)take((size_t)MM * 4);
  __bf16* la   = (__bf16*)take((size_t)MM * HHH * 2);
  __bf16* l2   = (__bf16*)take((size_t)MM * 128 * 2);
  __bf16* Ubf  = (__bf16*)take((size_t)MM * 128 * 2);
  __bf16* Wc1T  = (__bf16*)take((size_t)HHH * 128 * 2);
  __bf16* Wx2bT = (__bf16*)take((size_t)128 * 128 * 2);
  __bf16* Wl2bT = (__bf16*)take((size_t)128 * 128 * 2);

  // one-time prep (weights packed bf16, transposed for TDM 1-D staging)
  pack_cat_kernel<<<32, 256, 0, stream>>>(W_l_1a, W_l_1b, Wc1T);
  pack_kernel<<<64, 256, 0, stream>>>(W_x_2, Wx2bT);
  pack_kernel<<<64, 256, 0, stream>>>(W_l_2, Wl2bT);
  xa_kernel<<<MM / 256, 256, 0, stream>>>(x_a_state, assn_prev, action, xa);
  xbias_kernel<<<MM * 128 / 256, 256, 0, stream>>>(xa, W_x_1a, W_x_1b, Xab);
  initU_kernel<<<MM * 128 / 256, 256, 0, stream>>>(u_a0, u_b0, U);
  hipMemcpyAsync(G, gamma0, (size_t)MM * 128 * 4, hipMemcpyDeviceToDevice, stream);

  // presence attention + transposed softmax + loop-invariant A tensors
  presence_kernel<<<BB * NC, 256, 0, stream>>>(edge, avail, W1p, W2p, logit, dT);
  softmax_kernel<<<BB * NC, 256, 0, stream>>>(logit, Pin);
  buildA_kernel<<<MM, 256, 0, stream>>>(Pin, dT, W_emb_1a, W_emb_2, A1, A2);

  // T1 loop: l_a = einsum(A1, u_a);  [u_a|u_b] = relu(l_a @ [Wl1a|Wl1b] + Xab)
  for (int t = 0; t < 5; ++t) {
    einsum_kernel<<<MM, 256, 0, stream>>>(A1, U, la, HHH, 128);
    gemm_wmma_kernel<<<MM / 128, 256, 0, stream>>>(la, Wc1T, Xab, U, Ubf, HHH, 1, 1);
  }

  // x2 = u_concat @ W_x_2 (no bias / no relu), A = bf16 shadow of U
  gemm_wmma_kernel<<<MM / 128, 256, 0, stream>>>(Ubf, Wx2bT, nullptr, X2, nullptr, 128, 0, 0);

  // T2 loop: l_2 = einsum(A2, gamma);  gamma = relu(l_2 @ W_l_2 + x2)
  for (int t = 0; t < 5; ++t) {
    einsum_kernel<<<MM, 256, 0, stream>>>(A2, G, l2, 128, 128);
    gemm_wmma_kernel<<<MM / 128, 256, 0, stream>>>(l2, Wl2bT, X2, G, nullptr, 128, 1, 1);
  }

  // Q[b] = sum_n gamma @ W_Q
  q_kernel<<<BB, 256, 0, stream>>>(G, W_Q, (float*)d_out);
}
